// CrossAttention_61847529062960
// MI455X (gfx1250) — matrix operand order repaired and checked by reading, not compiled, not run on hardware
//
#include <hip/hip_runtime.h>

#define DIMD     1024
#define NHEADS   16
#define HEADDIM  64
#define BATCH    4
#define SEQ      2048
#define SCALE_F  0.125f   // 1/sqrt(64)

typedef __bf16 bf16;
typedef __attribute__((ext_vector_type(16))) __bf16 bf16x16;
typedef __attribute__((ext_vector_type(8)))  __bf16 bf16x8;
typedef __attribute__((ext_vector_type(8)))  float  f32x8;

enum { EPI_Q = 0, EPI_KV = 1, EPI_OUT = 2 };

static __device__ __forceinline__ int lane_id() { return (int)(threadIdx.x & 31u); }

static __device__ __forceinline__ f32x8 wmma_bf16(bf16x16 a, bf16x16 b, f32x8 c) {
  return __builtin_amdgcn_wmma_f32_16x16x32_bf16(false, a, false, b, (short)0, c, false, false);
}

// A fragment (16x32 bf16, row-major source, leading dim lda).
// Lane L: row = L%16, K = half*8 + e + (e>=8 ? 8 : 0)  -> two contiguous 16B loads.
static __device__ __forceinline__ bf16x16 load_afrag(const bf16* __restrict__ A, int lda, int k0) {
  int lane = lane_id();
  int half = lane >> 4;
  int r = lane & 15;
  const bf16* p = A + (size_t)r * lda + k0 + half * 8;
  bf16x8 lo = *(const bf16x8*)(p);
  bf16x8 hi = *(const bf16x8*)(p + 16);
  bf16x16 out;
#pragma unroll
  for (int e = 0; e < 8; ++e) { out[e] = lo[e]; out[e + 8] = hi[e]; }
  return out;
}

// B fragment (32x16 bf16) from B^T storage [N, ldb] (original column-major).
// Lane L: col = L%16, K = half*8 + e + (e>=8 ? 8 : 0)  -> two contiguous 16B loads.
static __device__ __forceinline__ bf16x16 load_bfrag(const bf16* __restrict__ Bt, int ldb, int k0) {
  int lane = lane_id();
  int half = lane >> 4;
  int c = lane & 15;
  const bf16* p = Bt + (size_t)c * ldb + k0 + half * 8;
  bf16x8 lo = *(const bf16x8*)(p);
  bf16x8 hi = *(const bf16x8*)(p + 16);
  bf16x16 out;
#pragma unroll
  for (int e = 0; e < 8; ++e) { out[e] = lo[e]; out[e + 8] = hi[e]; }
  return out;
}

// ---------------- conversion kernels ----------------
__global__ void cvt_bf16_kernel(const float* __restrict__ in, bf16* __restrict__ out, int n) {
  int i = blockIdx.x * blockDim.x + threadIdx.x;
  if (i < n) out[i] = (bf16)in[i];
}

// W[K][N] (row-major) -> Wt[N][K] bf16
__global__ void transpose_cvt_kernel(const float* __restrict__ W, bf16* __restrict__ Wt, int K, int N) {
  int i = blockIdx.x * blockDim.x + threadIdx.x;
  if (i < K * N) {
    int k = i / N;
    int n = i - k * N;
    Wt[(size_t)n * K + k] = (bf16)W[i];
  }
}

// ---------------- bf16 GEMM:  C[M,N] = A[M,K] @ Bt[N,K]^T + bias ----------------
// 256 threads = 8 waves (2 m x 4 n); wave tile 32x64; block tile 64x256.
// Software-pipelined: fragment loads for step k+1 issued before step k's WMMAs.
template<int EPI>
__global__ __launch_bounds__(256)
void gemm_bf16_kernel(const bf16* __restrict__ A, const bf16* __restrict__ Bt,
                      const float* __restrict__ bias,
                      void* __restrict__ out0, void* __restrict__ out1,
                      int K, int N) {
  int wave = threadIdx.x >> 5;
  int wm = wave & 1, wn = wave >> 1;
  int m0 = blockIdx.x * 64 + wm * 32;
  int n0 = blockIdx.y * 256 + wn * 64;
  int lane = lane_id();
  int half = lane >> 4;
  int cidx = lane & 15;
  (void)N;

  const bf16* A0 = A + (size_t)m0 * K;
  const bf16* A1 = A + (size_t)(m0 + 16) * K;

  const f32x8 zero = {0.f, 0.f, 0.f, 0.f, 0.f, 0.f, 0.f, 0.f};
  f32x8 acc[2][4];
#pragma unroll
  for (int i = 0; i < 2; ++i)
#pragma unroll
    for (int j = 0; j < 4; ++j) acc[i][j] = zero;

  // prologue: fragments for k0 = 0
  bf16x16 ca0 = load_afrag(A0, K, 0);
  bf16x16 ca1 = load_afrag(A1, K, 0);
  bf16x16 cb[4];
#pragma unroll
  for (int j = 0; j < 4; ++j) cb[j] = load_bfrag(Bt + (size_t)(n0 + j * 16) * K, K, 0);

  for (int k0 = 0; k0 < K; k0 += 32) {
    // prefetch next K-step's fragments before computing on the current ones
    bf16x16 na0 = ca0, na1 = ca1, nb[4] = {cb[0], cb[1], cb[2], cb[3]};
    int kn = k0 + 32;
    if (kn < K) {
      na0 = load_afrag(A0, K, kn);
      na1 = load_afrag(A1, K, kn);
#pragma unroll
      for (int j = 0; j < 4; ++j) nb[j] = load_bfrag(Bt + (size_t)(n0 + j * 16) * K, K, kn);
    }
#pragma unroll
    for (int j = 0; j < 4; ++j) {
      acc[0][j] = wmma_bf16(ca0, cb[j], acc[0][j]);
      acc[1][j] = wmma_bf16(ca1, cb[j], acc[1][j]);
    }
    ca0 = na0; ca1 = na1;
#pragma unroll
    for (int j = 0; j < 4; ++j) cb[j] = nb[j];
  }

  float bj[4];
#pragma unroll
  for (int j = 0; j < 4; ++j) bj[j] = bias[n0 + j * 16 + cidx];

#pragma unroll
  for (int i = 0; i < 2; ++i) {
#pragma unroll
    for (int j = 0; j < 4; ++j) {
#pragma unroll
      for (int g = 0; g < 8; ++g) {
        int row = m0 + i * 16 + half * 8 + g;     // C layout: M = g + half*8
        int col = n0 + j * 16 + cidx;             // N = lane%16
        float v = acc[i][j][g] + bj[j];
        if (EPI == EPI_Q) {
          int b = row >> 11, l = row & 2047;
          int h = col >> 6, hd = col & 63;
          ((bf16*)out0)[(((size_t)b * NHEADS + h) * SEQ + l) * HEADDIM + hd] = (bf16)v;
        } else if (EPI == EPI_KV) {
          int b = row >> 11, l = row & 2047;
          if (col < DIMD) {                        // K -> [B,H,L,Hd]
            int h = col >> 6, hd = col & 63;
            ((bf16*)out0)[(((size_t)b * NHEADS + h) * SEQ + l) * HEADDIM + hd] = (bf16)v;
          } else {                                 // V -> transposed [B,H,Hd,L]
            int c2 = col - DIMD;
            int h = c2 >> 6, hd = c2 & 63;
            ((bf16*)out1)[(((size_t)b * NHEADS + h) * HEADDIM + hd) * SEQ + l] = (bf16)v;
          }
        } else {                                   // EPI_OUT: fp32 final
          ((float*)out0)[(size_t)row * DIMD + col] = v;
        }
      }
    }
  }
}

// ---------------- flash attention ----------------
// grid(B*H, SEQ/64), 128 threads = 4 waves, one 16-row q-tile per wave.
// Pipelined: V fragments issued at loop top (consumed after softmax), next
// kv-tile's K fragments prefetched while the score WMMAs execute.
__global__ __launch_bounds__(128)
void flash_attn_kernel(const bf16* __restrict__ q, const bf16* __restrict__ k,
                       const bf16* __restrict__ vt, bf16* __restrict__ o) {
  __shared__ float S[4][16][32];
  __shared__ float rowfac[4][16];

  int wave = threadIdx.x >> 5;
  int lane = lane_id();
  int half = lane >> 4;
  int cidx = lane & 15;
  int bh = blockIdx.x;
  int b = bh >> 4, h = bh & 15;
  int q0 = blockIdx.y * 64 + wave * 16;

  const bf16* Q  = q  + ((size_t)bh * SEQ + q0) * HEADDIM;
  const bf16* Kh = k  + (size_t)bh * SEQ * HEADDIM;
  const bf16* Vh = vt + (size_t)bh * HEADDIM * SEQ;

  bf16x16 qa0 = load_afrag(Q, HEADDIM, 0);    // hd 0..31
  bf16x16 qa1 = load_afrag(Q, HEADDIM, 32);   // hd 32..63

  const f32x8 zero = {0.f, 0.f, 0.f, 0.f, 0.f, 0.f, 0.f, 0.f};
  f32x8 Oacc[4] = {zero, zero, zero, zero};
  float m_r = -3.0e38f, l_r = 0.0f;           // valid in lanes 0..15 (row = lane)

  // prologue: K fragments for kv tile 0.  kf[t*2+c]: subtile t (kv 0..15 /
  // 16..31), hd chunk c (0..31 / 32..63)
  bf16x16 kf[4];
#pragma unroll
  for (int t = 0; t < 2; ++t)
#pragma unroll
    for (int c = 0; c < 2; ++c)
      kf[t * 2 + c] = load_bfrag(Kh + (size_t)t * 16 * HEADDIM, HEADDIM, c * 32);

  for (int kv0 = 0; kv0 < SEQ; kv0 += 32) {
    // V fragments for this tile: issue early, consumed after softmax
    bf16x16 vf[4];
#pragma unroll
    for (int t = 0; t < 4; ++t)
      vf[t] = load_bfrag(Vh + (size_t)t * 16 * SEQ, SEQ, kv0);

    // S = Q @ K^T with preloaded fragments
    f32x8 s0 = zero, s1 = zero;
    s0 = wmma_bf16(qa0, kf[0], s0);
    s0 = wmma_bf16(qa1, kf[1], s0);
    s1 = wmma_bf16(qa0, kf[2], s1);
    s1 = wmma_bf16(qa1, kf[3], s1);

    // prefetch next kv tile's K fragments while softmax runs
    int kvn = kv0 + 32;
    if (kvn < SEQ) {
      const bf16* Kn = Kh + (size_t)kvn * HEADDIM;
#pragma unroll
      for (int t = 0; t < 2; ++t)
#pragma unroll
        for (int c = 0; c < 2; ++c)
          kf[t * 2 + c] = load_bfrag(Kn + (size_t)t * 16 * HEADDIM, HEADDIM, c * 32);
    }

#pragma unroll
    for (int g = 0; g < 8; ++g) {
      S[wave][half * 8 + g][cidx]      = s0[g] * SCALE_F;
      S[wave][half * 8 + g][16 + cidx] = s1[g] * SCALE_F;
    }
    asm volatile("s_wait_dscnt 0" ::: "memory");   // same-wave LDS RAW

    if (lane < 16) {                               // online softmax, row = lane
      float* Sr = S[wave][lane];
      float tmax = Sr[0];
#pragma unroll
      for (int j = 1; j < 32; ++j) tmax = fmaxf(tmax, Sr[j]);
      float nm  = fmaxf(m_r, tmax);
      float fac = __expf(m_r - nm);
      float sum = 0.f;
#pragma unroll
      for (int j = 0; j < 32; ++j) { float p = __expf(Sr[j] - nm); Sr[j] = p; sum += p; }
      l_r = l_r * fac + sum;
      m_r = nm;
      rowfac[wave][lane] = fac;
    }
    asm volatile("s_wait_dscnt 0" ::: "memory");

    float rs[8];
#pragma unroll
    for (int g = 0; g < 8; ++g) rs[g] = rowfac[wave][half * 8 + g];
#pragma unroll
    for (int t = 0; t < 4; ++t)
#pragma unroll
      for (int g = 0; g < 8; ++g) Oacc[t][g] *= rs[g];

    // P as bf16 A-fragment (16x32): row = lane%16, K = half*8 + e + (e>=8?8:0)
    bf16x16 pa;
#pragma unroll
    for (int e = 0; e < 16; ++e) {
      int kk = half * 8 + e + ((e >= 8) ? 8 : 0);
      pa[e] = (bf16)S[wave][cidx][kk];
    }
    // O += P @ V  (V fragments issued at loop top)
#pragma unroll
    for (int t = 0; t < 4; ++t) Oacc[t] = wmma_bf16(pa, vf[t], Oacc[t]);
  }

  if (lane < 16) rowfac[wave][lane] = 1.0f / l_r;
  asm volatile("s_wait_dscnt 0" ::: "memory");
  float rs[8];
#pragma unroll
  for (int g = 0; g < 8; ++g) rs[g] = rowfac[wave][half * 8 + g];
#pragma unroll
  for (int t = 0; t < 4; ++t)
#pragma unroll
    for (int g = 0; g < 8; ++g) {
      int qrow = q0 + half * 8 + g;
      o[((size_t)b * SEQ + qrow) * DIMD + h * HEADDIM + t * 16 + cidx] =
          (bf16)(Oacc[t][g] * rs[g]);
    }
}

// ---------------- host launcher ----------------
extern "C" void kernel_launch(void* const* d_in, const int* in_sizes, int n_in,
                              void* d_out, int out_size, void* d_ws, size_t ws_size,
                              hipStream_t stream) {
  const float* x   = (const float*)d_in[0];
  const float* ctx = (const float*)d_in[1];
  const float* Wq  = (const float*)d_in[2];
  const float* bq  = (const float*)d_in[3];
  const float* Wkv = (const float*)d_in[4];
  const float* bkv = (const float*)d_in[5];
  const float* Wo  = (const float*)d_in[6];
  const float* bo  = (const float*)d_in[7];
  float* out = (float*)d_out;
  (void)in_sizes; (void)n_in; (void)out_size; (void)ws_size;

  char* ws = (char*)d_ws;
  size_t off = 0;
  auto alloc = [&](size_t bytes) {
    char* p = ws + off;
    off += (bytes + 255) & ~(size_t)255;
    return p;
  };
  const size_t MQ = (size_t)BATCH * SEQ;                       // 8192 rows
  bf16* xb   = (bf16*)alloc(MQ * DIMD * sizeof(bf16));         // x bf16
  bf16* cb   = (bf16*)alloc(MQ * DIMD * sizeof(bf16));         // context bf16
  bf16* Wqt  = (bf16*)alloc((size_t)DIMD * DIMD * sizeof(bf16));
  bf16* Wkvt = (bf16*)alloc((size_t)2 * DIMD * DIMD * sizeof(bf16));
  bf16* Wot  = (bf16*)alloc((size_t)DIMD * DIMD * sizeof(bf16));
  bf16* qb   = (bf16*)alloc(MQ * DIMD * sizeof(bf16));         // [B,H,L,Hd]
  bf16* kb   = (bf16*)alloc(MQ * DIMD * sizeof(bf16));         // [B,H,L,Hd]
  bf16* vtb  = (bf16*)alloc(MQ * DIMD * sizeof(bf16));         // [B,H,Hd,L]
  bf16* ab   = (bf16*)alloc(MQ * DIMD * sizeof(bf16));         // attn out [B*L, D]

  const int nact = (int)(MQ * DIMD);                           // 8.39M
  cvt_bf16_kernel<<<(nact + 255) / 256, 256, 0, stream>>>(x, xb, nact);
  cvt_bf16_kernel<<<(nact + 255) / 256, 256, 0, stream>>>(ctx, cb, nact);
  transpose_cvt_kernel<<<(DIMD * DIMD + 255) / 256, 256, 0, stream>>>(Wq, Wqt, DIMD, DIMD);
  transpose_cvt_kernel<<<(2 * DIMD * DIMD + 255) / 256, 256, 0, stream>>>(Wkv, Wkvt, DIMD, 2 * DIMD);
  transpose_cvt_kernel<<<(DIMD * DIMD + 255) / 256, 256, 0, stream>>>(Wo, Wot, DIMD, DIMD);

  dim3 blk(256);
  gemm_bf16_kernel<EPI_Q><<<dim3((unsigned)(MQ / 64), DIMD / 256), blk, 0, stream>>>(
      xb, Wqt, bq, qb, nullptr, DIMD, DIMD);
  gemm_bf16_kernel<EPI_KV><<<dim3((unsigned)(MQ / 64), 2 * DIMD / 256), blk, 0, stream>>>(
      cb, Wkvt, bkv, kb, vtb, DIMD, 2 * DIMD);
  flash_attn_kernel<<<dim3(BATCH * NHEADS, SEQ / 64), 128, 0, stream>>>(qb, kb, vtb, ab);
  gemm_bf16_kernel<EPI_OUT><<<dim3((unsigned)(MQ / 64), DIMD / 256), blk, 0, stream>>>(
      ab, Wot, bo, out, nullptr, DIMD, DIMD);
}
